// CRAN_42477226557478
// MI455X (gfx1250) — matrix operand order repaired and checked by reading, not compiled, not run on hardware
//
#include <hip/hip_runtime.h>
#include <hip/hip_bf16.h>
#include <math.h>

// ---------------- problem constants ----------------
#define S_LEN 2048
#define BATCH 64
#define EMB   256
#define HID   256
#define G3H   768      // 3*HID
#define TWOH  512      // 2*HID
#define KN    256
#define OUTN  10

typedef __attribute__((ext_vector_type(16))) __bf16 v16bf;
typedef __attribute__((ext_vector_type(8)))  float  v8f;
typedef unsigned int u32x4 __attribute__((ext_vector_type(4)));
typedef int          i32x4 __attribute__((ext_vector_type(4)));
typedef int          i32x8 __attribute__((ext_vector_type(8)));

// ---------------- WMMA helpers (CDNA5 16x16x32 bf16) ----------------
__device__ __forceinline__ v8f wmma_bf16(v16bf a, v16bf b, v8f c) {
  return __builtin_amdgcn_wmma_f32_16x16x32_bf16(
      /*neg_a=*/false, a, /*neg_b=*/false, b,
      /*c_mod=*/(short)0, c, /*reuse_a=*/false, /*reuse_b=*/false);
}

// A tile 16x32 bf16 from row-major src (row stride ld elements).
__device__ __forceinline__ v16bf load_a_16x32(const __bf16* src, int ld) {
  int lane = threadIdx.x & 31;
  int hi   = (lane >> 4) ? 16 : 0;
  int m    = lane & 15;
  v16bf a;
#pragma unroll
  for (int j = 0; j < 8; ++j) {
    int k = 2 * j + hi;
    a[2 * j]     = src[m * ld + k];
    a[2 * j + 1] = src[m * ld + k + 1];
  }
  return a;
}

// B tile 32x16 where B[k][n] = W[(n0+n)*ld + k0+k] (W stored so k is contiguous
// per lane -> vectorizes into 2 x b128 loads per lane)
__device__ __forceinline__ v16bf load_b_32x16_T(const __bf16* W, int ld, int n0, int k0) {
  int lane = threadIdx.x & 31;
  int n    = lane & 15;
  int kh   = (lane >> 4) ? 16 : 0;
  v16bf b;
#pragma unroll
  for (int j = 0; j < 16; ++j) b[j] = W[(n0 + n) * ld + k0 + kh + j];
  return b;
}

// Sliding-window conv B tile: B[k][n] = emb_lds[(p0+n + k/256)*256 + (k%256)]
__device__ __forceinline__ v16bf load_b_conv(const __bf16* embs, int k0, int p0) {
  int lane = threadIdx.x & 31;
  int n    = lane & 15;
  int kh   = (lane >> 4) ? 16 : 0;
  v16bf b;
#pragma unroll
  for (int j = 0; j < 16; ++j) {
    int k   = k0 + kh + j;
    int i   = k >> 8;
    int e   = k & 255;
    int row = p0 + n + i;
    b[j] = (row < BATCH) ? embs[row * 256 + e] : (__bf16)0.0f;
  }
  return b;
}

// ---------------- TDM: async 1-D tile copy global -> LDS ----------------
// D# per CDNA5 ISA: group0 = {count/flags, lds_addr,
// global_addr lo, global_addr hi | type=2<<30}; group1 carries data_size=2B,
// tensor_dim0 = tile_dim0 = nelem, stride = nelem.
__device__ __forceinline__ void tdm_load_bf16(const __bf16* gsrc, __bf16* lds_dst,
                                              unsigned nelem /* bf16 elems, <=65535 */) {
  unsigned long long ga = (unsigned long long)(size_t)gsrc;
  unsigned lds = (unsigned)(size_t)lds_dst;       // generic addr low 32 = LDS offset
  u32x4 g0;
  g0[0] = 1u;                                      // count=1, user descriptor
  g0[1] = lds;                                     // lds_addr
  g0[2] = (unsigned)(ga & 0xFFFFFFFFu);            // global_addr[31:0]
  g0[3] = (unsigned)((ga >> 32) & 0x01FFFFFFu)     // global_addr[56:32]
          | 0x80000000u;                           // type=2 in bits[127:126]
  i32x8 g1;
  g1[0] = 0x00010000;                              // data_size=1 (2 bytes)
  g1[1] = (int)(nelem << 16);                      // tensor_dim0[15:0]
  g1[2] = (int)((nelem >> 16) | (1u << 16));       // tensor_dim0[31:16], tensor_dim1=1
  g1[3] = (int)(nelem << 16);                      // tile_dim0
  g1[4] = 0;                                       // tile_dim1/2 unused
  g1[5] = (int)nelem;                              // tensor_dim0_stride[31:0]
  g1[6] = 0;
  g1[7] = 0;
  i32x4 z = {0, 0, 0, 0};
#if defined(__clang_major__) && (__clang_major__ >= 23)
  i32x8 z8 = {0, 0, 0, 0, 0, 0, 0, 0};
  __builtin_amdgcn_tensor_load_to_lds(g0, g1, z, z, z8, 0);
#else
  __builtin_amdgcn_tensor_load_to_lds(g0, g1, z, z, 0);
#endif
}

// ---------------- prep kernels ----------------
__global__ void k_cast_bf16(__bf16* dst, const float* src, int n) {
  int i = blockIdx.x * blockDim.x + threadIdx.x;
  if (i < n) dst[i] = (__bf16)src[i];
}

// cast + transpose (n x n) f32 -> bf16: dst[c*n + r] = src[r*n + c]
__global__ void k_cast_bf16_T512(__bf16* dst, const float* src) {
  int idx = blockIdx.x * blockDim.x + threadIdx.x;   // over 512*512
  int r = idx >> 9, c = idx & 511;
  dst[c * 512 + r] = (__bf16)src[r * 512 + c];
}

// pack conv weight (KN,E,k) row-major -> A[o][i*E+e] bf16
__global__ void k_pack_conv(__bf16* dst, const float* src, int ksz) {
  int n = KN * EMB * ksz;
  int idx = blockIdx.x * blockDim.x + threadIdx.x;
  if (idx >= n) return;
  int K = EMB * ksz;
  int o = idx / K;
  int r = idx - o * K;
  int i = r >> 8;
  int e = r & 255;
  dst[idx] = (__bf16)src[(o * EMB + e) * ksz + i];
}

// embedding gather -> bf16 (S,B,E)
__global__ void k_embed(__bf16* emb_bf, const int* ids, const float* lookup) {
  int sb = blockIdx.x;
  int e  = threadIdx.x;
  int id = ids[sb];
  emb_bf[(size_t)sb * EMB + e] = (__bf16)lookup[(size_t)id * EMB + e];
}

// ---------------- GRU: one workgroup per direction, persistent scan ----------------
// LDS: h 32K | rz(f32) 128K | ghn(bf16) 32K | gxn(bf16) 32K | emb dbl-buf 64K = 288K
__global__ __launch_bounds__(1024)
void k_gru(const __bf16* __restrict__ emb_bf,
           const __bf16* __restrict__ Wih_f, const __bf16* __restrict__ Whh_f,
           const float* __restrict__ bih_f,  const float* __restrict__ bhh_f,
           const __bf16* __restrict__ Wih_b, const __bf16* __restrict__ Whh_b,
           const float* __restrict__ bih_b,  const float* __restrict__ bhh_b,
           const float* __restrict__ state_word,
           __bf16* __restrict__ ow_bf) {
  extern __shared__ char smem[];
  __bf16* h_bf   = (__bf16*)smem;                              // 32768
  float*  buf_rz = (float*)(smem + 32768);                     // 131072
  __bf16* ghn    = (__bf16*)(smem + 163840);                   // 32768
  __bf16* gxn    = (__bf16*)(smem + 196608);                   // 32768
  __bf16* embl   = (__bf16*)(smem + 229376);                   // 2 * 32768

  const int dir = blockIdx.x;
  const int tid = threadIdx.x;
  const int wid = tid >> 5;

  const __bf16* Wih = dir ? Wih_b : Wih_f;
  const __bf16* Whh = dir ? Whh_b : Whh_f;
  const float*  bih = dir ? bih_b : bih_f;
  const float*  bhh = dir ? bhh_b : bhh_f;

  const float* h0 = state_word + (size_t)dir * BATCH * HID;
  for (int i = tid; i < BATCH * HID; i += 1024) h_bf[i] = (__bf16)h0[i];

  // kick off TDM prefetch of the first timestep's emb slice (wave 0 only)
  const int s0 = dir ? (S_LEN - 1) : 0;
  if (wid == 0)
    tdm_load_bf16(emb_bf + (size_t)s0 * BATCH * EMB, embl, BATCH * EMB);
  __syncthreads();

  for (int it = 0; it < S_LEN; ++it) {
    const int s_in = dir ? (S_LEN - 1 - it) : it;
    const __bf16* emb_cur = embl + (it & 1) * (BATCH * EMB);

    // wave0: prefetch next step, then retire the current transfer.
    if (wid == 0) {
      if (it + 1 < S_LEN) {
        const int s_nx = dir ? (S_LEN - 2 - it) : (it + 1);
        tdm_load_bf16(emb_bf + (size_t)s_nx * BATCH * EMB,
                      embl + ((it + 1) & 1) * (BATCH * EMB), BATCH * EMB);
        __builtin_amdgcn_s_wait_tensorcnt(1);   // in-order: current tile done
      } else {
        __builtin_amdgcn_s_wait_tensorcnt(0);
      }
    }
    __syncthreads();   // publish emb_cur to all waves

    // 192 tiles (4 M x 48 N over 64x768), 32 waves -> 6 tiles each
#pragma unroll 1
    for (int t = wid * 6; t < wid * 6 + 6; ++t) {
      const int mt = t & 3;
      const int nt = t >> 2;
      v8f cgh = {}; v8f cgx = {};
#pragma unroll
      for (int kt = 0; kt < 8; ++kt) {     // h @ W_hh^T
        v16bf a = load_a_16x32(&h_bf[(mt * 16) * HID + kt * 32], HID);
        v16bf b = load_b_32x16_T(Whh, HID, nt * 16, kt * 32);
        cgh = wmma_bf16(a, b, cgh);
      }
#pragma unroll
      for (int kt = 0; kt < 8; ++kt) {     // x_t @ W_ih^T (x from LDS via TDM)
        v16bf a = load_a_16x32(&emb_cur[(mt * 16) * EMB + kt * 32], EMB);
        v16bf b = load_b_32x16_T(Wih, EMB, nt * 16, kt * 32);
        cgx = wmma_bf16(a, b, cgx);
      }
      const int lane = tid & 31;
      const int n    = lane & 15;
      const int mb   = (lane >> 4) ? 8 : 0;
      const int col  = nt * 16 + n;
      if (nt < 32) {                        // r/z region: sum both halves
        float bias = bhh[col] + bih[col];
#pragma unroll
        for (int r = 0; r < 8; ++r) {
          int row = mt * 16 + mb + r;
          buf_rz[row * TWOH + col] = cgh[r] + cgx[r] + bias;
        }
      } else {                              // n region: keep gh / gx separate
        int j = col - TWOH;
        float bh = bhh[col], bi = bih[col];
#pragma unroll
        for (int r = 0; r < 8; ++r) {
          int row = mt * 16 + mb + r;
          ghn[row * HID + j] = (__bf16)(cgh[r] + bh);
          gxn[row * HID + j] = (__bf16)(cgx[r] + bi);
        }
      }
    }
    __syncthreads();

    // gate math + hidden update + emit output
    for (int i = tid; i < BATCH * HID; i += 1024) {
      int b = i >> 8, j = i & 255;
      float pr = buf_rz[b * TWOH + j];
      float pz = buf_rz[b * TWOH + HID + j];
      float r  = 1.0f / (1.0f + expf(-pr));
      float z  = 1.0f / (1.0f + expf(-pz));
      float nn = tanhf((float)gxn[i] + r * (float)ghn[i]);
      float ho = (float)h_bf[i];
      float hn = (1.0f - z) * nn + z * ho;
      h_bf[i] = (__bf16)hn;
      ow_bf[((size_t)s_in * BATCH + b) * TWOH + dir * HID + j] = (__bf16)hn;
    }
    __syncthreads();
  }
}

// ---------------- CNN branch: conv3/4/5 as GEMM + relu/max + linear ----------------
__global__ __launch_bounds__(256)
void k_cnn(const __bf16* __restrict__ emb_bf,
           const __bf16* __restrict__ A3, const __bf16* __restrict__ A4,
           const __bf16* __restrict__ A5,
           const float* __restrict__ cb3, const float* __restrict__ cb4,
           const float* __restrict__ cb5,
           const float* __restrict__ lw, const float* __restrict__ lb,
           float* __restrict__ wproj) {
  extern __shared__ char smem[];
  __bf16* embs   = (__bf16*)smem;                 // 32KB
  float*  feats  = (float*)(smem + 32768);        // 768 f32
  int*    featsi = (int*)feats;

  const int s   = blockIdx.x;
  const int tid = threadIdx.x;
  const int wid = tid >> 5;

  for (int i = tid; i < BATCH * EMB; i += 256)
    embs[i] = emb_bf[(size_t)s * BATCH * EMB + i];
  for (int i = tid; i < 3 * KN; i += 256) featsi[i] = 0;
  __syncthreads();

  for (int t = wid; t < 192; t += 8) {
    const __bf16* A; const float* cb; int Ktiles, ksz, fbase, loc;
    if (t < 64)       { A = A3; cb = cb3; Ktiles = 24; ksz = 3; fbase = 0;      loc = t; }
    else if (t < 128) { A = A4; cb = cb4; Ktiles = 32; ksz = 4; fbase = KN;     loc = t - 64; }
    else              { A = A5; cb = cb5; Ktiles = 40; ksz = 5; fbase = 2 * KN; loc = t - 128; }
    const int mt = loc & 15;
    const int nt = loc >> 4;
    const int K  = Ktiles * 32;
    v8f acc = {};
#pragma unroll 1
    for (int kt = 0; kt < Ktiles; ++kt) {
      v16bf a = load_a_16x32(&A[(mt * 16) * K + kt * 32], K);
      v16bf b = load_b_conv(embs, kt * 32, nt * 16);
      acc = wmma_bf16(a, b, acc);
    }
    const int lane = tid & 31;
    const int n    = lane & 15;
    const int mb   = (lane >> 4) ? 8 : 0;
    const int p    = nt * 16 + n;
    const int P    = BATCH - ksz + 1;
    if (p < P) {
#pragma unroll
      for (int r = 0; r < 8; ++r) {
        int o = mt * 16 + mb + r;
        float v = acc[r] + cb[o];
        v = v > 0.0f ? v : 0.0f;
        atomicMax(&featsi[fbase + o], __float_as_int(v));
      }
    }
  }
  __syncthreads();

  for (int j = tid; j < TWOH; j += 256) {
    float a = lb[j];
    const float* w = &lw[(size_t)j * (3 * KN)];
    for (int c = 0; c < 3 * KN; ++c) a += feats[c] * w[c];
    wproj[(size_t)s * TWOH + j] = a;
  }
}

// ---------------- attention: squish = tanh(ow @ W_word + b); attn = <squish, wproj> ----------------
// WwordT is bf16, transposed (column-major) so B fragments vectorize.
__global__ __launch_bounds__(256)
void k_attn(const __bf16* __restrict__ ow_bf, const __bf16* __restrict__ WwordT,
            const float* __restrict__ b_word, const float* __restrict__ wproj,
            float* __restrict__ attn) {
  extern __shared__ float sq[];   // 64*512 f32 = 128KB
  const int s   = blockIdx.x;
  const int tid = threadIdx.x;
  const int wid = tid >> 5;
  const __bf16* ows = ow_bf + (size_t)s * BATCH * TWOH;

  for (int t = wid; t < 128; t += 8) {
    const int mt = t & 3;
    const int nt = t >> 2;
    v8f acc = {};
#pragma unroll
    for (int kt = 0; kt < 16; ++kt) {
      v16bf a = load_a_16x32(&ows[(mt * 16) * TWOH + kt * 32], TWOH);
      v16bf b = load_b_32x16_T(WwordT, TWOH, nt * 16, kt * 32);
      acc = wmma_bf16(a, b, acc);
    }
    const int lane = tid & 31;
    const int n    = lane & 15;
    const int mb   = (lane >> 4) ? 8 : 0;
    const int col  = nt * 16 + n;
    const float bw = b_word[col];
#pragma unroll
    for (int r = 0; r < 8; ++r)
      sq[(mt * 16 + mb + r) * TWOH + col] = tanhf(acc[r] + bw);
  }
  __syncthreads();

  if (tid < BATCH) {
    float a = 0.0f;
    const float* wp = &wproj[(size_t)s * TWOH];
    for (int k = 0; k < TWOH; ++k) a += sq[tid * TWOH + k] * wp[k];
    attn[(size_t)tid * S_LEN + s] = a;    // (B, S) for column softmax
  }
}

// ---------------- softmax over S per batch column (in-place) ----------------
__global__ __launch_bounds__(256)
void k_softmax_s(float* __restrict__ attn) {
  __shared__ float red[256];
  const int b   = blockIdx.x;
  const int tid = threadIdx.x;
  float* col = attn + (size_t)b * S_LEN;

  float m = -1e30f;
  for (int s = tid; s < S_LEN; s += 256) m = fmaxf(m, col[s]);
  red[tid] = m; __syncthreads();
  for (int o = 128; o > 0; o >>= 1) { if (tid < o) red[tid] = fmaxf(red[tid], red[tid + o]); __syncthreads(); }
  m = red[0]; __syncthreads();

  float sum = 0.0f;
  for (int s = tid; s < S_LEN; s += 256) { float e = expf(col[s] - m); col[s] = e; sum += e; }
  red[tid] = sum; __syncthreads();
  for (int o = 128; o > 0; o >>= 1) { if (tid < o) red[tid] += red[tid + o]; __syncthreads(); }
  float inv = 1.0f / red[0];
  for (int s = tid; s < S_LEN; s += 256) col[s] *= inv;
}

// ---------------- fc + output softmax ----------------
__global__ __launch_bounds__(256)
void k_fc(const __bf16* __restrict__ ow_bf, const float* __restrict__ attn,
          const float* __restrict__ fcw, const float* __restrict__ fcb,
          float* __restrict__ out) {
  __shared__ float part[256 * OUTN];
  __shared__ float red[OUTN];
  const int s   = blockIdx.x;
  const int tid = threadIdx.x;
  const int N   = BATCH * TWOH;

  float acc[OUTN];
#pragma unroll
  for (int o = 0; o < OUTN; ++o) acc[o] = 0.0f;

  for (int i = tid; i < N; i += 256) {
    int b = i >> 9;
    float w = (float)ow_bf[(size_t)s * N + i] * attn[(size_t)b * S_LEN + s];
#pragma unroll
    for (int o = 0; o < OUTN; ++o) acc[o] += w * fcw[(size_t)o * N + i];
  }
#pragma unroll
  for (int o = 0; o < OUTN; ++o) part[tid * OUTN + o] = acc[o];
  __syncthreads();

  if (tid < OUTN) {
    float v = fcb[tid];
    for (int j = 0; j < 256; ++j) v += part[j * OUTN + tid];
    red[tid] = v;
  }
  __syncthreads();
  if (tid == 0) {
    float m = -1e30f;
    for (int o = 0; o < OUTN; ++o) m = fmaxf(m, red[o]);
    float sum = 0.0f;
    float e[OUTN];
    for (int o = 0; o < OUTN; ++o) { e[o] = expf(red[o] - m); sum += e[o]; }
    float inv = 1.0f / sum;
    for (int o = 0; o < OUTN; ++o) out[(size_t)s * OUTN + o] = e[o] * inv;
  }
}

// ---------------- host launcher ----------------
extern "C" void kernel_launch(void* const* d_in, const int* in_sizes, int n_in,
                              void* d_out, int out_size, void* d_ws, size_t ws_size,
                              hipStream_t stream) {
  const int*   ids        = (const int*)  d_in[0];
  const float* state_word = (const float*)d_in[1];
  const float* lookup     = (const float*)d_in[2];
  const float* Wih_f      = (const float*)d_in[3];
  const float* Whh_f      = (const float*)d_in[4];
  const float* bih_f      = (const float*)d_in[5];
  const float* bhh_f      = (const float*)d_in[6];
  const float* Wih_b      = (const float*)d_in[7];
  const float* Whh_b      = (const float*)d_in[8];
  const float* bih_b      = (const float*)d_in[9];
  const float* bhh_b      = (const float*)d_in[10];
  const float* W_word     = (const float*)d_in[11];
  const float* b_word     = (const float*)d_in[12];
  const float* cw3        = (const float*)d_in[13];
  const float* cb3        = (const float*)d_in[14];
  const float* cw4        = (const float*)d_in[15];
  const float* cb4        = (const float*)d_in[16];
  const float* cw5        = (const float*)d_in[17];
  const float* cb5        = (const float*)d_in[18];
  const float* lw         = (const float*)d_in[19];
  const float* lb         = (const float*)d_in[20];
  const float* fcw        = (const float*)d_in[21];
  const float* fcb        = (const float*)d_in[22];

  char* ws = (char*)d_ws;
  size_t off = 0;
  auto carve = [&](size_t bytes) -> char* {
    char* p = ws + off;
    off += (bytes + 255) & ~(size_t)255;
    return p;
  };
  __bf16* emb_bf   = (__bf16*)carve((size_t)S_LEN * BATCH * EMB * 2);
  __bf16* ow_bf    = (__bf16*)carve((size_t)S_LEN * BATCH * TWOH * 2);
  __bf16* Wih_f_bf = (__bf16*)carve((size_t)G3H * EMB * 2);
  __bf16* Whh_f_bf = (__bf16*)carve((size_t)G3H * HID * 2);
  __bf16* Wih_b_bf = (__bf16*)carve((size_t)G3H * EMB * 2);
  __bf16* Whh_b_bf = (__bf16*)carve((size_t)G3H * HID * 2);
  __bf16* WwordT   = (__bf16*)carve((size_t)TWOH * TWOH * 2);
  __bf16* A3       = (__bf16*)carve((size_t)KN * EMB * 3 * 2);
  __bf16* A4       = (__bf16*)carve((size_t)KN * EMB * 4 * 2);
  __bf16* A5       = (__bf16*)carve((size_t)KN * EMB * 5 * 2);
  float*  wproj    = (float*) carve((size_t)S_LEN * TWOH * 4);
  float*  attn     = (float*) carve((size_t)S_LEN * BATCH * 4);

  auto cast = [&](__bf16* dst, const float* src, int n) {
    k_cast_bf16<<<(n + 255) / 256, 256, 0, stream>>>(dst, src, n);
  };
  cast(Wih_f_bf, Wih_f, G3H * EMB);
  cast(Whh_f_bf, Whh_f, G3H * HID);
  cast(Wih_b_bf, Wih_b, G3H * EMB);
  cast(Whh_b_bf, Whh_b, G3H * HID);
  k_cast_bf16_T512<<<(TWOH * TWOH + 255) / 256, 256, 0, stream>>>(WwordT, W_word);
  k_pack_conv<<<(KN * EMB * 3 + 255) / 256, 256, 0, stream>>>(A3, cw3, 3);
  k_pack_conv<<<(KN * EMB * 4 + 255) / 256, 256, 0, stream>>>(A4, cw4, 4);
  k_pack_conv<<<(KN * EMB * 5 + 255) / 256, 256, 0, stream>>>(A5, cw5, 5);
  k_embed<<<S_LEN * BATCH, EMB, 0, stream>>>(emb_bf, ids, lookup);

  // GRU: 2 persistent workgroups (fwd/bwd), 288KB dynamic LDS, TDM double-buffer
  k_gru<<<2, 1024, 294912, stream>>>(emb_bf,
                                     Wih_f_bf, Whh_f_bf, bih_f, bhh_f,
                                     Wih_b_bf, Whh_b_bf, bih_b, bhh_b,
                                     state_word, ow_bf);

  // CNN branch
  k_cnn<<<S_LEN, 256, 32768 + 3 * KN * 4, stream>>>(emb_bf, A3, A4, A5,
                                                    cb3, cb4, cb5, lw, lb, wproj);

  // attention projection + scores
  k_attn<<<S_LEN, 256, BATCH * TWOH * 4, stream>>>(ow_bf, WwordT, b_word, wproj, attn);

  // softmax over sequence dim per batch column
  k_softmax_s<<<BATCH, 256, 0, stream>>>(attn);

  // weighted fc + output softmax
  k_fc<<<S_LEN, 256, 0, stream>>>(ow_bf, attn, fcw, fcb, (float*)d_out);
}
